// DifferentialAttention_40776419508679
// MI455X (gfx1250) — matrix operand order repaired and checked by reading, not compiled, not run on hardware
//
#include <hip/hip_runtime.h>
#include <hip/hip_bf16.h>

typedef __attribute__((ext_vector_type(16))) __bf16 v16bf;
typedef __attribute__((ext_vector_type(8)))  float  v8f;

#define BM 128
#define BN 64
#define BK 32
#define LDSS 40   // LDS row stride in bf16 elems: 80 B rows -> every 16B chunk aligned

__device__ __forceinline__ __bf16 f2bf(float x) {
    unsigned b = __builtin_bit_cast(unsigned, x);
    b += 0x7FFFu + ((b >> 16) & 1u);           // round-to-nearest-even
    unsigned short h = (unsigned short)(b >> 16);
    return __builtin_bit_cast(__bf16, h);
}

// Async global -> LDS copy of 16 bytes (CDNA5 GLOBAL_LOAD_ASYNC_TO_LDS_B128).
__device__ __forceinline__ void async_g2l_b128(unsigned lds_off, const void* gptr) {
    asm volatile("global_load_async_to_lds_b128 %0, %1, off"
                 :: "v"(lds_off), "v"(gptr) : "memory");
}
__device__ __forceinline__ void wait_async0() {
    asm volatile("s_wait_asynccnt 0x0" ::: "memory");
}

// C[M,N] = alpha*(A@B) (+ bias).  A,B are bf16.  BT: B stored [N,K] (C = A@B^T).
// OBF: write bf16 C, else fp32 C.  HB: add fp32 bias[n].
// Double-buffered LDS pipeline: stage tile k+1 (async / ds) while computing tile k.
template <bool BT, bool OBF, bool HB>
__global__ __launch_bounds__(256) void gemm_wmma(
    const __bf16* __restrict__ A, const __bf16* __restrict__ B,
    const float* __restrict__ bias, void* __restrict__ Cv,
    int K, int lda, int ldb, int ldc,
    long long strideA, long long strideB, long long strideC, float alpha)
{
    __shared__ __bf16 ldsA[2 * BM * LDSS];
    __shared__ __bf16 ldsB[2 * BN * LDSS];

    const __bf16* Ab = A + (long long)blockIdx.z * strideA;
    const __bf16* Bb = B + (long long)blockIdx.z * strideB;

    const int tid  = threadIdx.x;
    const int lane = tid & 31;
    const int wave = tid >> 5;
    const int row0 = blockIdx.y * BM;
    const int col0 = blockIdx.x * BN;
    const int r    = lane & 15;   // row (A/C) or col (B) within 16-tile
    const int h    = lane >> 4;   // lane half

    auto stage = [&](int k0, __bf16* la, __bf16* lb) {
        // A tile [BM x BK]: async copy, 16B per transfer, 2 per thread
        #pragma unroll
        for (int i = 0; i < 2; ++i) {
            int chunk = tid + i * 256;          // 0..511 = 128 rows x 4 chunks
            int rr = chunk >> 2;
            int kc = (chunk & 3) * 8;
            async_g2l_b128((unsigned)(size_t)&la[rr * LDSS + kc],
                           Ab + (size_t)(row0 + rr) * lda + (k0 + kc));
        }
        if (BT) {
            // B is [N,K]: contiguous along k -> async copy, 1 chunk per thread
            int nn = tid >> 2;
            int kc = (tid & 3) * 8;
            async_g2l_b128((unsigned)(size_t)&lb[nn * LDSS + kc],
                           Bb + (size_t)(col0 + nn) * ldb + (k0 + kc));
        } else {
            // B is [K,N]: batch 4 independent dword loads (one clause, one wait),
            // then transpose into LDS via 8 ds_store_b16.
            unsigned vals[4];
            #pragma unroll
            for (int i = 0; i < 4; ++i) {
                int idx = tid + i * 256;        // 0..1023 dwords = 32 k x 32 dword-cols
                int kk = idx >> 5;
                int nn = (idx & 31) * 2;
                vals[i] = *(const unsigned*)(Bb + (size_t)(k0 + kk) * ldb + (col0 + nn));
            }
            #pragma unroll
            for (int i = 0; i < 4; ++i) {
                int idx = tid + i * 256;
                int kk = idx >> 5;
                int nn = (idx & 31) * 2;
                unsigned short lo = (unsigned short)(vals[i] & 0xffffu);
                unsigned short hi = (unsigned short)(vals[i] >> 16);
                lb[(nn + 0) * LDSS + kk] = __builtin_bit_cast(__bf16, lo);
                lb[(nn + 1) * LDSS + kk] = __builtin_bit_cast(__bf16, hi);
            }
        }
    };

    v8f acc[4] = {};

    stage(0, ldsA, ldsB);     // prologue: stage tile 0 into buffer 0
    int cur = 0;

    for (int k0 = 0; k0 < K; k0 += BK) {
        wait_async0();        // staging of 'cur' buffer complete (this wave)
        __syncthreads();      // ... and all waves' staging visible

        __bf16* la = ldsA + cur * (BM * LDSS);
        __bf16* lb = ldsB + cur * (BN * LDSS);

        // issue staging of next tile into the other buffer; overlaps with WMMA below
        if (k0 + BK < K)
            stage(k0 + BK, ldsA + (cur ^ 1) * (BM * LDSS), ldsB + (cur ^ 1) * (BN * LDSS));

        // A fragment per ISA 16-bit 16x32 layout: VGPR p, half h -> k = (p<4 ? 2p : 2p+8) + 8h
        union { v16bf v; unsigned u[8]; } fa;
        const __bf16* arow = &la[(wave * 16 + r) * LDSS];
        #pragma unroll
        for (int p = 0; p < 8; ++p) {
            int kb = (p < 4 ? 2 * p : 2 * p + 8) + 8 * h;
            fa.u[p] = *(const unsigned*)(arow + kb);
        }
        #pragma unroll
        for (int t = 0; t < 4; ++t) {
            // B fragment per ISA 16-bit 32x16 layout: VGPR p, half h -> k = (p<4 ? 2p : 2p+8) + 8h
            union { v16bf v; unsigned u[8]; } fb;
            const __bf16* bcol = &lb[(t * 16 + r) * LDSS];
            #pragma unroll
            for (int p = 0; p < 8; ++p) {
                int kb = (p < 4 ? 2 * p : 2 * p + 8) + 8 * h;
                fb.u[p] = *(const unsigned*)(bcol + kb);
            }
            acc[t] = __builtin_amdgcn_wmma_f32_16x16x32_bf16(
                false, fa.v, false, fb.v, (short)0, acc[t], false, false);
        }
        cur ^= 1;
    }

    // C/D layout: VGPR rr, lane -> M = rr + 8h, N = lane&15
    #pragma unroll
    for (int t = 0; t < 4; ++t) {
        #pragma unroll
        for (int rr = 0; rr < 8; ++rr) {
            int gm = row0 + wave * 16 + rr + 8 * h;
            int gn = col0 + t * 16 + r;
            float v = acc[t][rr] * alpha;
            if (HB) v += bias[gn];
            size_t cidx = (size_t)gm * ldc + gn;
            if (OBF) {
                __bf16* C = (__bf16*)Cv + (long long)blockIdx.z * strideC;
                C[cidx] = f2bf(v);
            } else {
                float* C = (float*)Cv + (long long)blockIdx.z * strideC;
                C[cidx] = v;
            }
        }
    }
}

__global__ __launch_bounds__(256) void cvt_bf16_kernel(const float* __restrict__ src,
                                                       __bf16* __restrict__ dst)
{
    int i = (blockIdx.x * 256 + threadIdx.x) * 4;
    float4 f = *(const float4*)(src + i);
    dst[i + 0] = f2bf(f.x);
    dst[i + 1] = f2bf(f.y);
    dst[i + 2] = f2bf(f.z);
    dst[i + 3] = f2bf(f.w);
}

__global__ void lam_kernel(const float* __restrict__ lq1, const float* __restrict__ lk1,
                           const float* __restrict__ lq2, const float* __restrict__ lk2,
                           float* __restrict__ out)
{
    if (threadIdx.x == 0) {
        float d1 = 0.f, d2 = 0.f;
        for (int i = 0; i < 64; ++i) {
            d1 += lq1[i] * lk1[i];
            d2 += lq2[i] * lk2[i];
        }
        // LAMBDA_INIT - 0.6*exp(-0.3*LAYER_INDEX), LAYER_INDEX = 0
        *out = __expf(d1) - __expf(d2) + (0.8f - 0.6f * __expf(-0.3f * 0.0f));
    }
}

// Per row (b,q): softmax both fp32 score rows (len 2048), write bf16 diff = p1 - lam*p2.
__global__ __launch_bounds__(256) void softmax_diff_kernel(
    const float* __restrict__ s1, const float* __restrict__ s2,
    __bf16* __restrict__ diff, const float* __restrict__ lamp)
{
    const int row = blockIdx.x;
    const int tid = threadIdx.x;
    const float* r1 = s1 + (size_t)row * 2048;
    const float* r2 = s2 + (size_t)row * 2048;
    __bf16*      rd = diff + (size_t)row * 2048;
    const float lam = *lamp;

    float a[8], b[8];
    float m1 = -3.0e38f, m2 = -3.0e38f;
    #pragma unroll
    for (int i = 0; i < 8; ++i) {
        a[i] = r1[tid + i * 256];
        b[i] = r2[tid + i * 256];
        m1 = fmaxf(m1, a[i]);
        m2 = fmaxf(m2, b[i]);
    }
    __shared__ float sm[256];
    sm[tid] = m1; __syncthreads();
    for (int s = 128; s > 0; s >>= 1) { if (tid < s) sm[tid] = fmaxf(sm[tid], sm[tid + s]); __syncthreads(); }
    m1 = sm[0]; __syncthreads();
    sm[tid] = m2; __syncthreads();
    for (int s = 128; s > 0; s >>= 1) { if (tid < s) sm[tid] = fmaxf(sm[tid], sm[tid + s]); __syncthreads(); }
    m2 = sm[0]; __syncthreads();

    float sum1 = 0.f, sum2 = 0.f;
    #pragma unroll
    for (int i = 0; i < 8; ++i) {
        a[i] = __expf(a[i] - m1);
        b[i] = __expf(b[i] - m2);
        sum1 += a[i];
        sum2 += b[i];
    }
    sm[tid] = sum1; __syncthreads();
    for (int s = 128; s > 0; s >>= 1) { if (tid < s) sm[tid] += sm[tid + s]; __syncthreads(); }
    sum1 = sm[0]; __syncthreads();
    sm[tid] = sum2; __syncthreads();
    for (int s = 128; s > 0; s >>= 1) { if (tid < s) sm[tid] += sm[tid + s]; __syncthreads(); }
    sum2 = sm[0];

    const float inv1 = 1.f / sum1;
    const float inv2 = lam / sum2;
    #pragma unroll
    for (int i = 0; i < 8; ++i)
        rd[tid + i * 256] = f2bf(a[i] * inv1 - b[i] * inv2);
}

extern "C" void kernel_launch(void* const* d_in, const int* in_sizes, int n_in,
                              void* d_out, int out_size, void* d_ws, size_t ws_size,
                              hipStream_t stream)
{
    const float* x    = (const float*)d_in[0];
    const float* Wqkv = (const float*)d_in[1];
    const float* bqkv = (const float*)d_in[2];
    const float* Wv   = (const float*)d_in[3];
    const float* bv   = (const float*)d_in[4];
    const float* Wo   = (const float*)d_in[5];
    const float* bo   = (const float*)d_in[6];
    const float* lq1  = (const float*)d_in[7];
    const float* lk1  = (const float*)d_in[8];
    const float* lq2  = (const float*)d_in[9];
    const float* lk2  = (const float*)d_in[10];
    float* out = (float*)d_out;

    const int Bb = 2, S = 2048, D = 512, QKV = 1024, VD = 8192;
    const int M = Bb * S;   // 4096

    // workspace layout: fp32 region then bf16 region (all 16B aligned)
    char*  w    = (char*)d_ws;
    float* lam  = (float*)w;
    float* sc1  = (float*)(w + 256);
    float* sc2  = sc1 + (size_t)Bb * S * S;           // 8,388,608 floats each
    __bf16* xbf    = (__bf16*)(sc2 + (size_t)Bb * S * S);
    __bf16* wqkvbf = xbf    + (size_t)M * D;
    __bf16* wvbf   = wqkvbf + (size_t)D * QKV;
    __bf16* wobf   = wvbf   + (size_t)D * VD;
    __bf16* qkvbf  = wobf   + (size_t)VD * D;
    __bf16* vbf    = qkvbf  + (size_t)M * QKV;
    __bf16* diffbf = vbf    + (size_t)M * VD;
    __bf16* aoutbf = diffbf + (size_t)Bb * S * S;

    lam_kernel<<<1, 64, 0, stream>>>(lq1, lk1, lq2, lk2, lam);

    // fp32 -> bf16 operand conversion (all sizes multiples of 1024)
    cvt_bf16_kernel<<<(M * D) / 1024,   256, 0, stream>>>(x,    xbf);
    cvt_bf16_kernel<<<(D * QKV) / 1024, 256, 0, stream>>>(Wqkv, wqkvbf);
    cvt_bf16_kernel<<<(D * VD) / 1024,  256, 0, stream>>>(Wv,   wvbf);
    cvt_bf16_kernel<<<(VD * D) / 1024,  256, 0, stream>>>(Wo,   wobf);

    // qkv = x @ Wqkv + bqkv  -> bf16        [4096,512]x[512,1024]
    gemm_wmma<false, true, true><<<dim3(QKV / BN, M / BM, 1), 256, 0, stream>>>(
        xbf, wqkvbf, bqkv, qkvbf, D, D, QKV, QKV, 0, 0, 0, 1.0f);

    // V = x @ Wv + bv  -> bf16              [4096,512]x[512,8192]
    gemm_wmma<false, true, true><<<dim3(VD / BN, M / BM, 1), 256, 0, stream>>>(
        xbf, wvbf, bv, vbf, D, D, VD, VD, 0, 0, 0, 1.0f);

    const float scale = 0.125f;  // 1/sqrt(HEAD_DIM=64)
    // scores1 = Q1 @ K1^T * scale -> fp32 (batched); Q1=qkv[:,0:256], K1=qkv[:,512:768]
    gemm_wmma<true, false, false><<<dim3(S / BN, S / BM, Bb), 256, 0, stream>>>(
        qkvbf + 0, qkvbf + 512, nullptr, sc1, 256, QKV, QKV, S,
        (long long)S * QKV, (long long)S * QKV, (long long)S * S, scale);
    // scores2 = Q2 @ K2^T * scale; Q2=qkv[:,256:512], K2=qkv[:,768:1024]
    gemm_wmma<true, false, false><<<dim3(S / BN, S / BM, Bb), 256, 0, stream>>>(
        qkvbf + 256, qkvbf + 768, nullptr, sc2, 256, QKV, QKV, S,
        (long long)S * QKV, (long long)S * QKV, (long long)S * S, scale);

    // diff = softmax(sc1) - lam * softmax(sc2) -> bf16
    softmax_diff_kernel<<<Bb * S, 256, 0, stream>>>(sc1, sc2, diffbf, lam);

    // aout = diff @ V -> bf16 (batched)      [2048,2048]x[2048,8192] per batch
    gemm_wmma<false, true, false><<<dim3(VD / BN, S / BM, Bb), 256, 0, stream>>>(
        diffbf, vbf, nullptr, aoutbf, S, S, VD, VD,
        (long long)S * S, (long long)S * VD, (long long)S * VD, 1.0f);

    // out = aout @ Wo + bo -> fp32           [4096,8192]x[8192,512]
    gemm_wmma<false, false, true><<<dim3(D / BN, M / BM, 1), 256, 0, stream>>>(
        aoutbf, wobf, bo, out, VD, VD, D, D, 0, 0, 0, 1.0f);
}
